// ThreeConv_14242111553629
// MI455X (gfx1250) — matrix operand order repaired and checked by reading, not compiled, not run on hardware
//
#include <hip/hip_runtime.h>
#include <hip/hip_bf16.h>
#include <math.h>

// CDNA5 / gfx1250: wave32, WMMA 16x16x32 bf16 -> f32
typedef __bf16 bf16_t;
typedef __attribute__((ext_vector_type(16))) __bf16 v16bf;
typedef __attribute__((ext_vector_type(8)))  float  v8f;

#define HEADS 4

// ---------------------------------------------------------------------------
// Edge kernel: one wave32 handles a tile of 16 edges.
//  1) gather xs/xd (fp32), stage xs as bf16 in LDS, compute attention logits
//  2) softmax over 4 heads -> q (fp32, LDS)
//  3) WMMA: [16 edges x KPAD] (bf16) @ [KPAD x 16-col slice of W] (bf16)
//  4) per-lane: acc += q[edge][head] * C; atomic scatter into num[dst]
// ---------------------------------------------------------------------------
template<int FIN, int OUT>
__global__ __launch_bounds__(256)
void feast_edge_kernel(const float* __restrict__ x,
                       const int*   __restrict__ ei,
                       const float* __restrict__ W,
                       const float* __restrict__ u,
                       const float* __restrict__ cvec,
                       float*       __restrict__ num,
                       int E, int N)
{
    constexpr int HOUT = HEADS * OUT;
    constexpr int KPAD = (FIN >= 32) ? FIN : 32;   // 64 / 32 / 32
    constexpr int NKS  = KPAD / 32;                // K-steps of 32
    constexpr int CH   = FIN / 2;                  // features per lane-half

    __shared__ bf16_t sW[HOUT * KPAD];             // transposed: [col][k], zero-padded K
    __shared__ float  sU[FIN * HEADS];
    __shared__ float  sC[HEADS];
    __shared__ bf16_t sX[8 * 16 * KPAD];           // per-wave 16xKPAD bf16 tiles
    __shared__ float  sQ[8 * 16 * HEADS];          // per-wave softmax weights
    __shared__ int    sD[8 * 16];                  // per-wave dst indices

    const int tid = threadIdx.x;

    // Preload W (transposed + K-padded), u, c into LDS
    for (int idx = tid; idx < HOUT * KPAD; idx += 256) {
        int col = idx / KPAD;
        int k   = idx - col * KPAD;
        sW[idx] = (k < FIN) ? (bf16_t)W[k * HOUT + col] : (bf16_t)0.0f;
    }
    for (int idx = tid; idx < FIN * HEADS; idx += 256) sU[idx] = u[idx];
    if (tid < HEADS) sC[tid] = cvec[tid];
    __syncthreads();

    const int wave = tid >> 5;
    const int lane = tid & 31;
    const int tile = blockIdx.x * 8 + wave;
    const int EN   = E + N;
    const int e16  = lane & 15;      // edge within tile
    const int hlf  = lane >> 4;      // which feature half this lane gathers

    int eg = tile * 16 + e16;
    bool valid = (eg < EN);
    int srcN = 0, dstN = 0;
    if (valid) {
        if (eg < E) { srcN = ei[eg]; dstN = ei[E + eg]; }
        else        { srcN = dstN = eg - E; }        // self loop
    }

    // Gather + partial attention dots: pd[h] = sum_k (xs-xd)*u[k][h] over this half
    float pd0 = 0.f, pd1 = 0.f, pd2 = 0.f, pd3 = 0.f;
    const int kb = hlf * CH;
    const float* xsrow = x + (size_t)srcN * FIN + kb;
    const float* xdrow = x + (size_t)dstN * FIN + kb;
    bf16_t* myx = &sX[(wave * 16 + e16) * KPAD];
    #pragma unroll
    for (int k = 0; k < CH; ++k) {
        float xs = xsrow[k];
        float xd = xdrow[k];
        myx[kb + k] = (bf16_t)xs;
        float d = xs - xd;
        const float* ur = &sU[(kb + k) * HEADS];
        pd0 = fmaf(d, ur[0], pd0);
        pd1 = fmaf(d, ur[1], pd1);
        pd2 = fmaf(d, ur[2], pd2);
        pd3 = fmaf(d, ur[3], pd3);
    }
    if constexpr (KPAD > FIN) {                    // zero-pad K for WMMA
        constexpr int PADH = (KPAD - FIN) / 2;
        #pragma unroll
        for (int k = 0; k < PADH; ++k) myx[FIN + hlf * PADH + k] = (bf16_t)0.0f;
    }

    // Combine the two half-lanes of each edge, then softmax on lanes 0..15
    pd0 += __shfl_xor(pd0, 16);
    pd1 += __shfl_xor(pd1, 16);
    pd2 += __shfl_xor(pd2, 16);
    pd3 += __shfl_xor(pd3, 16);
    if (hlf == 0) {
        float l0 = pd0 + sC[0], l1 = pd1 + sC[1], l2 = pd2 + sC[2], l3 = pd3 + sC[3];
        float mx = fmaxf(fmaxf(l0, l1), fmaxf(l2, l3));
        float q0 = __expf(l0 - mx), q1 = __expf(l1 - mx);
        float q2 = __expf(l2 - mx), q3 = __expf(l3 - mx);
        float inv = valid ? 1.0f / (q0 + q1 + q2 + q3) : 0.0f;  // invalid edges -> q=0
        float* qrow = &sQ[(wave * 16 + e16) * HEADS];
        qrow[0] = q0 * inv; qrow[1] = q1 * inv; qrow[2] = q2 * inv; qrow[3] = q3 * inv;
        sD[wave * 16 + e16] = dstN;
    }
    __syncthreads();

    // ---- A fragments (16x32 bf16 tile per K-step), CDNA5 wave32 layout:
    // lane<16: elems 0..7 = K 0..7, elems 8..15 = K 16..23 of row=lane
    // lane>=16: elems 0..7 = K 8..15, elems 8..15 = K 24..31 of row=lane-16
    v16bf afrag[NKS];
    {
        const int row = lane & 15;
        const int kbA = hlf * 8;
        const bf16_t* xr = &sX[(wave * 16 + row) * KPAD];
        #pragma unroll
        for (int ks = 0; ks < NKS; ++ks) {
            #pragma unroll
            for (int j = 0; j < 8; ++j) {
                afrag[ks][j]     = xr[ks * 32 + kbA + j];
                afrag[ks][8 + j] = xr[ks * 32 + 16 + kbA + j];
            }
        }
    }

    const int colL  = lane & 15;       // output column within 16-wide tile
    const int kbB   = hlf * 8;         // B: lanes 0-15 hold K 0..7,16..23; lanes 16-31 K 8..15,24..31
    const int erow0 = hlf * 8;         // C: VGPR r -> row r + 8*(lane>>4)
    const float* qw = &sQ[wave * 16 * HEADS];
    const int*   dw = &sD[wave * 16];

    #pragma unroll
    for (int ot = 0; ot < OUT / 16; ++ot) {
        float acc[8];
        #pragma unroll
        for (int r = 0; r < 8; ++r) acc[r] = 0.0f;

        #pragma unroll
        for (int h = 0; h < HEADS; ++h) {
            const int gcol = h * OUT + ot * 16 + colL;
            const bf16_t* wcol = &sW[gcol * KPAD];
            v8f cfr = {};
            #pragma unroll
            for (int ks = 0; ks < NKS; ++ks) {
                v16bf bfrag;
                #pragma unroll
                for (int m = 0; m < 16; ++m) bfrag[m] = wcol[ks * 32 + (m >> 3) * 16 + kbB + (m & 7)];
                cfr = __builtin_amdgcn_wmma_f32_16x16x32_bf16(
                          /*neg_a=*/false, afrag[ks], /*neg_b=*/false, bfrag,
                          /*c_mod=*/(short)0, cfr, /*reuse_a=*/false, /*reuse_b=*/false);
            }
            #pragma unroll
            for (int r = 0; r < 8; ++r)
                acc[r] = fmaf(qw[(erow0 + r) * HEADS + h], cfr[r], acc[r]);
        }

        const int oc = ot * 16 + colL;
        #pragma unroll
        for (int r = 0; r < 8; ++r)
            atomicAdd(&num[(size_t)dw[erow0 + r] * OUT + oc], acc[r]);
    }
}

// ---------------------------------------------------------------------------
__global__ __launch_bounds__(256)
void deg_kernel(const int* __restrict__ ei, float* __restrict__ deg, int E, int N)
{
    int i = blockIdx.x * 256 + threadIdx.x;
    if (i >= E + N) return;
    int d = (i < E) ? ei[E + i] : (i - E);
    atomicAdd(&deg[d], 1.0f);
}

template<int OUT>
__global__ __launch_bounds__(256)
void finalize_kernel(float* __restrict__ h, const float* __restrict__ deg,
                     const float* __restrict__ b, int N)
{
    int idx = blockIdx.x * 256 + threadIdx.x;
    if (idx >= N * OUT) return;
    int i = idx / OUT;
    int c = idx - i * OUT;
    float v = h[idx] / fmaxf(deg[i], 1.0f) + b[c];
    h[idx] = fmaxf(v, 0.0f);               // ReLU (used after all three convs)
}

// BatchNorm: per-channel partial sums (thread = channel), then tiny finalize
__global__ __launch_bounds__(64)
void bn_stats_kernel(const float* __restrict__ h, float* __restrict__ stats, int N)
{
    int c = threadIdx.x;                   // 0..63
    float s = 0.f, s2 = 0.f;
    for (int i = blockIdx.x; i < N; i += gridDim.x) {
        float v = h[(size_t)i * 64 + c];
        s += v; s2 += v * v;
    }
    atomicAdd(&stats[c],      s);
    atomicAdd(&stats[64 + c], s2);
}

__global__ __launch_bounds__(64)
void bn_finalize_kernel(float* __restrict__ stats, const float* __restrict__ gamma,
                        const float* __restrict__ beta, int N)
{
    int c = threadIdx.x;
    float invn = 1.0f / (float)N;
    float mu  = stats[c] * invn;
    float var = stats[64 + c] * invn - mu * mu;
    float rs  = rsqrtf(var + 1e-5f);
    float g   = rs * gamma[c];
    stats[128 + c] = g;                    // scale
    stats[192 + c] = beta[c] - mu * g;     // shift
}

// MLP head: tiny (2.7 KFLOP/node), memory-bound -> scalar FMA, weights via L2
__global__ __launch_bounds__(256)
void mlp_kernel(const float* __restrict__ h, const float* __restrict__ stats,
                const float* __restrict__ lw1, const float* __restrict__ lb1,
                const float* __restrict__ lw2, const float* __restrict__ lb2,
                const float* __restrict__ lw3, const float* __restrict__ lb3,
                const float* __restrict__ lw4, const float* __restrict__ lb4,
                const float* __restrict__ ow,  const float* __restrict__ ob,
                float* __restrict__ out, int N)
{
    int i = blockIdx.x * 256 + threadIdx.x;
    if (i >= N) return;

    float a1[32];
    #pragma unroll
    for (int j = 0; j < 32; ++j) a1[j] = lb1[j];
    for (int c = 0; c < 64; ++c) {
        float v = fmaf(h[(size_t)i * 64 + c], stats[128 + c], stats[192 + c]); // BN
        #pragma unroll
        for (int j = 0; j < 32; ++j) a1[j] = fmaf(v, lw1[c * 32 + j], a1[j]);
    }
    float a2[16];
    #pragma unroll
    for (int j = 0; j < 16; ++j) a2[j] = lb2[j];
    #pragma unroll
    for (int c = 0; c < 32; ++c) {
        float v = fmaxf(a1[c], 0.0f);
        #pragma unroll
        for (int j = 0; j < 16; ++j) a2[j] = fmaf(v, lw2[c * 16 + j], a2[j]);
    }
    float a3[8];
    #pragma unroll
    for (int j = 0; j < 8; ++j) a3[j] = lb3[j];
    #pragma unroll
    for (int c = 0; c < 16; ++c) {
        float v = fmaxf(a2[c], 0.0f);
        #pragma unroll
        for (int j = 0; j < 8; ++j) a3[j] = fmaf(v, lw3[c * 8 + j], a3[j]);
    }
    float a4[4];
    #pragma unroll
    for (int j = 0; j < 4; ++j) a4[j] = lb4[j];
    #pragma unroll
    for (int c = 0; c < 8; ++c) {
        float v = fmaxf(a3[c], 0.0f);
        #pragma unroll
        for (int j = 0; j < 4; ++j) a4[j] = fmaf(v, lw4[c * 4 + j], a4[j]);
    }
    float o = ob[0];
    #pragma unroll
    for (int j = 0; j < 4; ++j) o = fmaf(fmaxf(a4[j], 0.0f), ow[j], o);
    out[i] = 1.0f / (1.0f + __expf(-o));
}

// ---------------------------------------------------------------------------
extern "C" void kernel_launch(void* const* d_in, const int* in_sizes, int n_in,
                              void* d_out, int out_size, void* d_ws, size_t ws_size,
                              hipStream_t stream)
{
    (void)n_in; (void)out_size; (void)ws_size;

    const float* x     = (const float*)d_in[0];
    const int*   ei    = (const int*)  d_in[1];
    const float* W1    = (const float*)d_in[2];
    const float* u1    = (const float*)d_in[3];
    const float* c1    = (const float*)d_in[4];
    const float* b1    = (const float*)d_in[5];
    const float* W2    = (const float*)d_in[6];
    const float* u2    = (const float*)d_in[7];
    const float* c2    = (const float*)d_in[8];
    const float* b2    = (const float*)d_in[9];
    const float* W3    = (const float*)d_in[10];
    const float* u3    = (const float*)d_in[11];
    const float* c3    = (const float*)d_in[12];
    const float* b3    = (const float*)d_in[13];
    const float* gamma = (const float*)d_in[14];
    const float* beta  = (const float*)d_in[15];
    const float* lw1   = (const float*)d_in[16];
    const float* lb1   = (const float*)d_in[17];
    const float* lw2   = (const float*)d_in[18];
    const float* lb2   = (const float*)d_in[19];
    const float* lw3   = (const float*)d_in[20];
    const float* lb3   = (const float*)d_in[21];
    const float* lw4   = (const float*)d_in[22];
    const float* lb4   = (const float*)d_in[23];
    const float* ow    = (const float*)d_in[24];
    const float* ob    = (const float*)d_in[25];

    const int N  = in_sizes[0] / 64;   // 50000
    const int E  = in_sizes[1] / 2;    // 800000
    const int EN = E + N;

    // Workspace: two N*64 ping-pong feature buffers + deg + BN stats (~26 MB)
    float* ws0   = (float*)d_ws;                 // N*64
    float* ws1   = ws0 + (size_t)N * 64;         // N*64
    float* deg   = ws1 + (size_t)N * 64;         // N
    float* stats = deg + N;                      // 256

    hipMemsetAsync(deg,   0, sizeof(float) * (size_t)N,       stream);
    hipMemsetAsync(stats, 0, sizeof(float) * 256,             stream);
    hipMemsetAsync(ws0,   0, sizeof(float) * (size_t)N * 16,  stream);

    deg_kernel<<<(EN + 255) / 256, 256, 0, stream>>>(ei, deg, E, N);

    const int tiles   = (EN + 15) / 16;
    const int eblocks = (tiles + 7) / 8;         // 8 waves (tiles) per block

    // Layer 1: 64 -> 16  (x -> ws0)
    feast_edge_kernel<64, 16><<<eblocks, 256, 0, stream>>>(x, ei, W1, u1, c1, ws0, E, N);
    finalize_kernel<16><<<(N * 16 + 255) / 256, 256, 0, stream>>>(ws0, deg, b1, N);

    // Layer 2: 16 -> 32  (ws0 -> ws1)
    hipMemsetAsync(ws1, 0, sizeof(float) * (size_t)N * 32, stream);
    feast_edge_kernel<16, 32><<<eblocks, 256, 0, stream>>>(ws0, ei, W2, u2, c2, ws1, E, N);
    finalize_kernel<32><<<(N * 32 + 255) / 256, 256, 0, stream>>>(ws1, deg, b2, N);

    // Layer 3: 32 -> 64  (ws1 -> ws0)
    hipMemsetAsync(ws0, 0, sizeof(float) * (size_t)N * 64, stream);
    feast_edge_kernel<32, 64><<<eblocks, 256, 0, stream>>>(ws1, ei, W3, u3, c3, ws0, E, N);
    finalize_kernel<64><<<(N * 64 + 255) / 256, 256, 0, stream>>>(ws0, deg, b3, N);

    // BatchNorm stats + MLP head -> d_out
    bn_stats_kernel<<<512, 64, 0, stream>>>(ws0, stats, N);
    bn_finalize_kernel<<<1, 64, 0, stream>>>(stats, gamma, beta, N);
    mlp_kernel<<<(N + 255) / 256, 256, 0, stream>>>(ws0, stats,
        lw1, lb1, lw2, lb2, lw3, lb3, lw4, lb4, ow, ob, (float*)d_out, N);
}
